// QSM_V4_19516331393145
// MI455X (gfx1250) — matrix operand order repaired and checked by reading, not compiled, not run on hardware
//
#include <hip/hip_runtime.h>

// ---------------- problem constants (match reference) ----------------
constexpr int cV = 32000, cD = 1024, cH = 16, cL = 3, cF = 4096, cB = 8, cS = 512;
constexpr int cDK = cD / cH;          // 64
constexpr int cNT = cB * cS;          // 4096 tokens

// ---------------- types ----------------
typedef __attribute__((ext_vector_type(16))) __bf16 v16bf;
typedef __attribute__((ext_vector_type(8)))  float  v8f;

union FragU { v16bf v; unsigned int w[8]; };

// ---------------- bf16 helpers (bit-exact RNE convert) ----------------
__device__ __forceinline__ unsigned short f2bf(float f) {
  unsigned int u = __float_as_uint(f);
  unsigned int r = (u + 0x7FFFu + ((u >> 16) & 1u)) >> 16;
  return (unsigned short)r;
}
__device__ __forceinline__ float bf2f(unsigned short h) {
  return __uint_as_float(((unsigned int)h) << 16);
}
// LDS byte offset of a shared-memory pointer (flat addr low 32 bits = LDS offset)
__device__ __forceinline__ unsigned lds_off(const void* p) {
  return (unsigned)(unsigned long long)p;
}

// =====================================================================
// GEMM: C(f32,opt) / Cb(bf16,opt) = act(alpha * A(bf16)[M,K] x B(bf16)[N,K]^T + bias)
// B is ALWAYS [N,K] row-major (weights pre-transposed on conversion).
// FULL=1: M%128==0, N%128==0, K%32==0 -> no guards, async-to-LDS double buffer.
// FULL=0: N may be < 128 but N%8==0, M%128==0, K%32==0 (vector-granular zero fill).
// ACT=1: exact GELU. CBT=1: bf16 mirror written transposed Cb[n*ldcb+m].
// Batched via blockIdx.z (strA/strB; strC offsets both C and Cb).
// Tiles: BM=128, BN=128, BK=32; 256 threads = 8 waves; wave tile 32x64.
// =====================================================================
template<int FULL, int ACT, int CBT>
__global__ __launch_bounds__(256)
void gemm_bt_wmma(const unsigned short* __restrict__ A, long long lda, long long strA,
                  const unsigned short* __restrict__ Bm, long long ldb, long long strB,
                  float* __restrict__ C, long long ldc, long long strC,
                  unsigned short* __restrict__ Cb, long long ldcb,
                  const float* __restrict__ bias,
                  int M, int N, int K, float alpha)
{
  __shared__ alignas(16) unsigned short sA[2][128 * 40];  // [buf][m][k] stride 40
  __shared__ alignas(16) unsigned short sB[2][128 * 40];  // [buf][n][k] stride 40

  const int tid    = threadIdx.x;
  const int lane   = tid & 31;
  const int wid    = tid >> 5;
  const int lane16 = lane & 15;
  const int laneh  = lane >> 4;
  const int wm     = wid & 3;     // 4 waves along M (32 rows each)
  const int wn     = wid >> 2;    // 2 waves along N (64 cols each)
  const long long bz = blockIdx.z;
  const int m0 = blockIdx.y * 128;
  const int n0 = blockIdx.x * 128;

  A  += bz * strA;
  Bm += bz * strB;
  C  = C  ? (C  + bz * strC) : C;
  Cb = Cb ? (Cb + bz * strC) : Cb;

  v8f acc[2][4];
#pragma unroll
  for (int i = 0; i < 2; ++i)
#pragma unroll
    for (int j = 0; j < 4; ++j)
#pragma unroll
      for (int r = 0; r < 8; ++r) acc[i][j][r] = 0.0f;

  const int nkt = (K + 31) >> 5;

  // ---- async tile fill (FULL path): 2 x b128 per thread per matrix ----
  auto issue = [&](int kt, int buf) {
    const int kb = kt << 5;
#pragma unroll
    for (int i = 0; i < 2; ++i) {
      int v   = tid + (i << 8);
      int row = v >> 2;
      int kq  = (v & 3) << 3;
      const unsigned short* ga = A + (long long)(m0 + row) * lda + (kb + kq);
      unsigned la = lds_off(&sA[buf][row * 40 + kq]);
      asm volatile("global_load_async_to_lds_b128 %0, %1, off"
                   :: "v"(la), "v"(ga) : "memory");
      const unsigned short* gb = Bm + (long long)(n0 + row) * ldb + (kb + kq);
      unsigned lb = lds_off(&sB[buf][row * 40 + kq]);
      asm volatile("global_load_async_to_lds_b128 %0, %1, off"
                   :: "v"(lb), "v"(gb) : "memory");
    }
  };

  // ---- ds_load fragments + 8 WMMA on a buffer ----
  auto compute = [&](int buf) {
    FragU af[2], bfr[4];
#pragma unroll
    for (int i = 0; i < 2; ++i) {
      // A 16x32 layout: lanes 0-15 rows, laneh k-half; w[0..3]=k[laneh*8..], w[4..7]=k[16+laneh*8..]
      const unsigned short* p = &sA[buf][(wm * 32 + i * 16 + lane16) * 40 + laneh * 8];
      uint4 lo = *(const uint4*)p;
      uint4 hi = *(const uint4*)(p + 16);
      af[i].w[0] = lo.x; af[i].w[1] = lo.y; af[i].w[2] = lo.z; af[i].w[3] = lo.w;
      af[i].w[4] = hi.x; af[i].w[5] = hi.y; af[i].w[6] = hi.z; af[i].w[7] = hi.w;
    }
#pragma unroll
    for (int j = 0; j < 4; ++j) {
      // B 32x16 layout: lane = column n, laneh k-half; w[0..3]=k[laneh*8..], w[4..7]=k[16+laneh*8..]
      const unsigned short* p = &sB[buf][(wn * 64 + j * 16 + lane16) * 40 + laneh * 8];
      uint4 lo = *(const uint4*)p;
      uint4 hi = *(const uint4*)(p + 16);
      bfr[j].w[0] = lo.x; bfr[j].w[1] = lo.y; bfr[j].w[2] = lo.z; bfr[j].w[3] = lo.w;
      bfr[j].w[4] = hi.x; bfr[j].w[5] = hi.y; bfr[j].w[6] = hi.z; bfr[j].w[7] = hi.w;
    }
#pragma unroll
    for (int i = 0; i < 2; ++i)
#pragma unroll
      for (int j = 0; j < 4; ++j)
        acc[i][j] = __builtin_amdgcn_wmma_f32_16x16x32_bf16(
            false, af[i].v, false, bfr[j].v, (short)0, acc[i][j], false, false);
  };

  if (FULL) {
    // double-buffered: async DMA of tile kt+1 overlaps WMMA on tile kt
    issue(0, 0);
    asm volatile("s_wait_asynccnt 0x0" ::: "memory");
    __syncthreads();
    for (int kt = 0; kt < nkt; ++kt) {
      const int cur = kt & 1;
      if (kt + 1 < nkt) issue(kt + 1, cur ^ 1);
      compute(cur);
      asm volatile("s_wait_asynccnt 0x0" ::: "memory");
      __syncthreads();
    }
  } else {
    // synchronous single-buffer path; vector-granular zero fill on B columns
    for (int kt = 0; kt < nkt; ++kt) {
      const int kb = kt << 5;
#pragma unroll
      for (int i = 0; i < 2; ++i) {
        int v   = tid + (i << 8);
        int row = v >> 2;
        int kq  = (v & 3) << 3;
        uint4 da = *(const uint4*)(A + (long long)(m0 + row) * lda + (kb + kq));
        *(uint4*)&sA[0][row * 40 + kq] = da;
        uint4 db;
        int gn = n0 + row;
        if (gn < N) db = *(const uint4*)(Bm + (long long)gn * ldb + (kb + kq));
        else { db.x = 0u; db.y = 0u; db.z = 0u; db.w = 0u; }
        *(uint4*)&sB[0][row * 40 + kq] = db;
      }
      __syncthreads();
      compute(0);
      __syncthreads();
    }
  }

  // ---- epilogue: C layout — lanes 0-15: M=r, lanes 16-31: M=r+8 ----
#pragma unroll
  for (int i = 0; i < 2; ++i) {
    int mb = m0 + wm * 32 + i * 16 + laneh * 8;
#pragma unroll
    for (int j = 0; j < 4; ++j) {
      int n = n0 + wn * 64 + j * 16 + lane16;
      float bv = 0.0f;
      if (bias && (FULL || n < N)) bv = bias[n];
#pragma unroll
      for (int r = 0; r < 8; ++r) {
        int m = mb + r;
        float val = acc[i][j][r] * alpha + bv;
        if (ACT == 1) val = 0.5f * val * (1.0f + erff(val * 0.70710678118654752f));
        if (FULL || (m < M && n < N)) {
          if (C)  C[(long long)m * ldc + n] = val;
          if (Cb) {
            if (CBT) Cb[(long long)n * ldcb + m] = f2bf(val);
            else     Cb[(long long)m * ldcb + n] = f2bf(val);
          }
        }
      }
    }
  }
}

// =====================================================================
// elementwise / reduction kernels
// =====================================================================
__global__ void cvt_bf16_kernel(const float* __restrict__ s, unsigned short* __restrict__ d,
                                long long n) {
  long long i = (long long)blockIdx.x * blockDim.x + threadIdx.x;
  long long st = (long long)gridDim.x * blockDim.x;
  for (; i < n; i += st) d[i] = f2bf(s[i]);
}

// transpose-convert: src f32 [K,N] -> dst bf16 [N,K]; K,N multiples of 32
__global__ __launch_bounds__(256)
void cvtT_kernel(const float* __restrict__ src, unsigned short* __restrict__ dst,
                 int K, int N) {
  __shared__ unsigned short t[32][33];
  const int tx = threadIdx.x & 31, ty = threadIdx.x >> 5;
  const int k0 = blockIdx.y << 5, n0 = blockIdx.x << 5;
#pragma unroll
  for (int i = 0; i < 4; ++i)
    t[ty + i * 8][tx] = f2bf(src[(long long)(k0 + ty + i * 8) * N + (n0 + tx)]);
  __syncthreads();
#pragma unroll
  for (int i = 0; i < 4; ++i)
    dst[(long long)(n0 + ty + i * 8) * K + (k0 + tx)] = t[tx][ty + i * 8];
}

__global__ __launch_bounds__(256)
void embed_kernel(const int* __restrict__ ids, const float* __restrict__ emb,
                  float* __restrict__ out, unsigned short* __restrict__ obf) {
  const int t = blockIdx.x;          // token index [0, NT)
  const int s = t & (cS - 1);        // position in sequence
  const long long id = ids[t];
  const float scale = 32.0f;         // sqrt(1024)
  for (int d = threadIdx.x; d < cD; d += 256) {
    int i2 = d & ~1;
    float div = __expf(-(float)i2 * (9.210340371976184f / (float)cD));
    float ang = (float)s * div;
    float pe = (d & 1) ? __cosf(ang) : __sinf(ang);
    float v = emb[id * cD + d] * scale + pe;
    out[(long long)t * cD + d] = v;
    obf[(long long)t * cD + d] = f2bf(v);
  }
}

__global__ __launch_bounds__(256)
void gate_kernel(const float* __restrict__ gate, float* __restrict__ out) {
  __shared__ float red[256];
  float s = 0.0f;
  for (int i = threadIdx.x; i < cD; i += 256) s += gate[i];
  red[threadIdx.x] = s; __syncthreads();
  for (int o = 128; o > 0; o >>= 1) {
    if (threadIdx.x < o) red[threadIdx.x] += red[threadIdx.x + o];
    __syncthreads();
  }
  if (threadIdx.x == 0) out[0] = 1.0f / (1.0f + __expf(-red[0] / (float)cD));
}

// fold gate+rotation into Q (bf16 in place): Qeff = g*rot(Q) + (1-g)*Q on ch 0,1
__global__ __launch_bounds__(256)
void qgate_kernel(unsigned short* __restrict__ qb, const float* __restrict__ theta,
                  const float* __restrict__ gscal) {
  long long idx = (long long)blockIdx.x * blockDim.x + threadIdx.x;  // token*H + h
  if (idx >= (long long)cNT * cH) return;
  int h = (int)(idx & (cH - 1));
  long long t = idx >> 4;
  float g  = gscal[0];
  float th = theta[h];
  float c = __cosf(th), sn = __sinf(th);
  long long base = t * cD + (long long)h * cDK;
  float q0 = bf2f(qb[base]), q1 = bf2f(qb[base + 1]);
  float r0 = c * q0 - sn * q1;
  float r1 = sn * q0 + c * q1;
  qb[base]     = f2bf(g * r0 + (1.0f - g) * q0);
  qb[base + 1] = f2bf(g * r1 + (1.0f - g) * q1);
}

// masked softmax over scores[h][q][k] for one batch element b -> bf16 probs
__global__ __launch_bounds__(256)
void softmax_mask_kernel(const float* __restrict__ sc, unsigned short* __restrict__ pb,
                         const int* __restrict__ ids, int b, int mode) {
  const int q = blockIdx.x;
  const int h = blockIdx.y;
  const long long off = ((long long)h * cS + q) * cS;
  const float* row = sc + off;
  unsigned short* orow = pb + off;
  __shared__ float red[256];
  const int tid = threadIdx.x;
  float v0, v1;
  {
    int k = tid;
    bool keep = (ids[b * cS + k] != 0);
    if (mode) keep = keep && (k <= q);
    v0 = keep ? row[k] : -1e9f;
    k = tid + 256;
    keep = (ids[b * cS + k] != 0);
    if (mode) keep = keep && (k <= q);
    v1 = keep ? row[k] : -1e9f;
  }
  float mx = fmaxf(v0, v1);
  red[tid] = mx; __syncthreads();
  for (int o = 128; o > 0; o >>= 1) {
    if (tid < o) red[tid] = fmaxf(red[tid], red[tid + o]);
    __syncthreads();
  }
  mx = red[0]; __syncthreads();
  v0 = __expf(v0 - mx);
  v1 = __expf(v1 - mx);
  red[tid] = v0 + v1; __syncthreads();
  for (int o = 128; o > 0; o >>= 1) {
    if (tid < o) red[tid] += red[tid + o];
    __syncthreads();
  }
  float inv = 1.0f / red[0];
  orow[tid]       = f2bf(v0 * inv);
  orow[tid + 256] = f2bf(v1 * inv);
}

// x = LN(x + delta) ; also emits bf16 mirror
__global__ __launch_bounds__(256)
void add_layernorm_kernel(float* __restrict__ x, const float* __restrict__ dlt,
                          const float* __restrict__ g, const float* __restrict__ be,
                          unsigned short* __restrict__ xbf) {
  const int t = blockIdx.x;
  float* px = x + (long long)t * cD;
  const float* pd = dlt + (long long)t * cD;
  __shared__ float red[256];
  const int tid = threadIdx.x;
  float loc[4];
  float s = 0.0f;
#pragma unroll
  for (int i = 0; i < 4; ++i) {
    int d = tid + i * 256;
    loc[i] = px[d] + pd[d];
    s += loc[i];
  }
  red[tid] = s; __syncthreads();
  for (int o = 128; o > 0; o >>= 1) { if (tid < o) red[tid] += red[tid + o]; __syncthreads(); }
  float mu = red[0] * (1.0f / (float)cD); __syncthreads();
  float s2 = 0.0f;
#pragma unroll
  for (int i = 0; i < 4; ++i) { float dv = loc[i] - mu; s2 += dv * dv; }
  red[tid] = s2; __syncthreads();
  for (int o = 128; o > 0; o >>= 1) { if (tid < o) red[tid] += red[tid + o]; __syncthreads(); }
  float inv = rsqrtf(red[0] * (1.0f / (float)cD) + 1e-5f);
#pragma unroll
  for (int i = 0; i < 4; ++i) {
    int d = tid + i * 256;
    float v = (loc[i] - mu) * inv * g[d] + be[d];
    px[d] = v;
    xbf[(long long)t * cD + d] = f2bf(v);
  }
}

// =====================================================================
// host orchestration
// =====================================================================
extern "C" void kernel_launch(void* const* d_in, const int* in_sizes, int n_in,
                              void* d_out, int out_size, void* d_ws, size_t ws_size,
                              hipStream_t stream) {
  (void)in_sizes; (void)out_size;
  if (n_in < 147) return;

  int idx = 0;
  const int* src_ids = (const int*)d_in[idx++];
  const int* tgt_ids = (const int*)d_in[idx++];
  const float* emb   = (const float*)d_in[idx++];

  struct AttnP { const float *Wq,*bq,*Wk,*bk,*Wv,*bv,*Wo,*bo,*theta,*gate; };
  struct FfnP  { const float *W1,*b1,*W2,*b2; };
  struct LnP   { const float *g,*b; };
  auto gfp = [&]() { return (const float*)d_in[idx++]; };
  auto gA  = [&]() { AttnP a; a.Wq=gfp(); a.bq=gfp(); a.Wk=gfp(); a.bk=gfp();
                     a.Wv=gfp(); a.bv=gfp(); a.Wo=gfp(); a.bo=gfp();
                     a.theta=gfp(); a.gate=gfp(); return a; };
  auto gFn = [&]() { FfnP f; f.W1=gfp(); f.b1=gfp(); f.W2=gfp(); f.b2=gfp(); return f; };
  auto gL  = [&]() { LnP l; l.g=gfp(); l.b=gfp(); return l; };

  AttnP encA[cL]; FfnP encF[cL]; LnP eL1[cL], eL2[cL];
  for (int l = 0; l < cL; ++l) { encA[l]=gA(); encF[l]=gFn(); eL1[l]=gL(); eL2[l]=gL(); }
  AttnP decSA[cL], decCA[cL]; FfnP decF[cL]; LnP dL1[cL], dL2[cL], dL3[cL];
  for (int l = 0; l < cL; ++l) { decSA[l]=gA(); decCA[l]=gA(); decF[l]=gFn();
                                 dL1[l]=gL(); dL2[l]=gL(); dL3[l]=gL(); }

  // ---- workspace carve ----
  char* wp = (char*)d_ws;
  auto af = [&](size_t n) { float* p = (float*)wp;
                            wp += ((n * sizeof(float) + 255) & ~(size_t)255); return p; };
  auto ah = [&](size_t n) { unsigned short* p = (unsigned short*)wp;
                            wp += ((n * 2 + 255) & ~(size_t)255); return p; };
  float*          xe    = af((size_t)cNT * cD);
  unsigned short* xe_bf = ah((size_t)cNT * cD);
  float*          yd    = af((size_t)cNT * cD);
  unsigned short* yd_bf = ah((size_t)cNT * cD);
  unsigned short* h1_bf = ah((size_t)cNT * cF);
  float*          qf    = af((size_t)cNT * cD);   // generic f32 delta / proj output
  unsigned short* qb    = ah((size_t)cNT * cD);
  unsigned short* kb    = ah((size_t)cNT * cD);
  unsigned short* vt    = ah((size_t)cD * cNT);   // V transposed: [D][NT]
  unsigned short* atb   = ah((size_t)cNT * cD);
  float*          sc    = af((size_t)cH * cS * cS);   // per-b scores
  unsigned short* pbuf  = ah((size_t)cH * cS * cS);
  unsigned short* wbuf  = ah((size_t)cV * cD);        // weight staging (bf16, [N,K])
  float*          gscal = af(64);
  if ((size_t)(wp - (char*)d_ws) > ws_size) return;

  auto cvt = [&](const float* s, unsigned short* dptr, long long n) {
    long long blk = (n + 255) / 256; if (blk > 16384) blk = 16384;
    cvt_bf16_kernel<<<(int)blk, 256, 0, stream>>>(s, dptr, n);
  };
  auto cvtT = [&](const float* w, int K, int N) {   // w [K,N] -> wbuf [N,K]
    cvtT_kernel<<<dim3(N / 32, K / 32), 256, 0, stream>>>(w, wbuf, K, N);
  };
  auto gemm = [&](const unsigned short* A, long long lda, long long strA,
                  const unsigned short* Bm, long long ldb, long long strB,
                  float* C, long long ldc, long long strC,
                  unsigned short* Cb, long long ldcb,
                  const float* bias, int M, int N, int K, int batch,
                  float alpha, int act, int full, int cbt) {
    dim3 grid((unsigned)((N + 127) / 128), (unsigned)((M + 127) / 128), (unsigned)batch);
    if (!full)
      gemm_bt_wmma<0, 0, 0><<<grid, 256, 0, stream>>>(A, lda, strA, Bm, ldb, strB,
                                                      C, ldc, strC, Cb, ldcb, bias, M, N, K, alpha);
    else if (cbt)
      gemm_bt_wmma<1, 0, 1><<<grid, 256, 0, stream>>>(A, lda, strA, Bm, ldb, strB,
                                                      C, ldc, strC, Cb, ldcb, bias, M, N, K, alpha);
    else if (act)
      gemm_bt_wmma<1, 1, 0><<<grid, 256, 0, stream>>>(A, lda, strA, Bm, ldb, strB,
                                                      C, ldc, strC, Cb, ldcb, bias, M, N, K, alpha);
    else
      gemm_bt_wmma<1, 0, 0><<<grid, 256, 0, stream>>>(A, lda, strA, Bm, ldb, strB,
                                                      C, ldc, strC, Cb, ldcb, bias, M, N, K, alpha);
  };

  const float iscale = 0.125f;  // 1/sqrt(DK)

  auto attn_block = [&](const unsigned short* xq_bf, const unsigned short* xkv_bf,
                        const AttnP& p, const int* mids, int mode) {
    cvtT(p.Wq, cD, cD);
    gemm(xq_bf, cD, 0, wbuf, cD, 0, nullptr, 0, 0, qb, cD, p.bq,
         cNT, cD, cD, 1, 1.0f, 0, 1, 0);
    cvtT(p.Wk, cD, cD);
    gemm(xkv_bf, cD, 0, wbuf, cD, 0, nullptr, 0, 0, kb, cD, p.bk,
         cNT, cD, cD, 1, 1.0f, 0, 1, 0);
    cvtT(p.Wv, cD, cD);
    gemm(xkv_bf, cD, 0, wbuf, cD, 0, nullptr, 0, 0, vt, cNT, p.bv,
         cNT, cD, cD, 1, 1.0f, 0, 1, 1);                 // bf16 mirror transposed -> [D][NT]
    gate_kernel<<<1, 256, 0, stream>>>(p.gate, gscal);
    qgate_kernel<<<(cNT * cH) / 256, 256, 0, stream>>>(qb, p.theta, gscal);
    for (int b = 0; b < cB; ++b) {
      const unsigned short* qbb = qb + (long long)b * cS * cD;
      const unsigned short* kbb = kb + (long long)b * cS * cD;
      // scores[h] = (Q_h @ K_h^T) / sqrt(DK); batch over heads via DK stride
      gemm(qbb, cD, cDK, kbb, cD, cDK,
           sc, cS, (long long)cS * cS, nullptr, 0, nullptr,
           cS, cS, cDK, cH, iscale, 0, 1, 0);
      softmax_mask_kernel<<<dim3(cS, cH), 256, 0, stream>>>(sc, pbuf, mids, b, mode);
      // attn_out[h] = probs[h] @ V_h ; B = vt slice [DK, S] per head (N=64 -> FULL=0)
      gemm(pbuf, cS, (long long)cS * cS,
           vt + (long long)b * cS, cNT, (long long)cDK * cNT,
           nullptr, 0, cDK, atb + (long long)b * cS * cD, cD, nullptr,
           cS, cDK, cS, cH, 1.0f, 0, 0, 0);
    }
    cvtT(p.Wo, cD, cD);
    gemm(atb, cD, 0, wbuf, cD, 0, qf, cD, 0, nullptr, 0, p.bo,
         cNT, cD, cD, 1, 1.0f, 0, 1, 0);
  };

  auto ffn_block = [&](const unsigned short* x_bf, const FfnP& p) {
    cvtT(p.W1, cD, cF);
    gemm(x_bf, cD, 0, wbuf, cD, 0, nullptr, 0, 0, h1_bf, cF, p.b1,
         cNT, cF, cD, 1, 1.0f, 1, 1, 0);                 // fused exact GELU
    cvtT(p.W2, cF, cD);
    gemm(h1_bf, cF, 0, wbuf, cF, 0, qf, cD, 0, nullptr, 0, p.b2,
         cNT, cD, cF, 1, 1.0f, 0, 1, 0);
  };
  auto addln = [&](float* x, unsigned short* xbf, const LnP& l) {
    add_layernorm_kernel<<<cNT, 256, 0, stream>>>(x, qf, l.g, l.b, xbf);
  };

  // ---------------- encoder ----------------
  embed_kernel<<<cNT, 256, 0, stream>>>(src_ids, emb, xe, xe_bf);
  for (int l = 0; l < cL; ++l) {
    attn_block(xe_bf, xe_bf, encA[l], src_ids, 0);
    addln(xe, xe_bf, eL1[l]);
    ffn_block(xe_bf, encF[l]);
    addln(xe, xe_bf, eL2[l]);
  }
  // ---------------- decoder ----------------
  embed_kernel<<<cNT, 256, 0, stream>>>(tgt_ids, emb, yd, yd_bf);
  for (int l = 0; l < cL; ++l) {
    attn_block(yd_bf, yd_bf, decSA[l], tgt_ids, 1);
    addln(yd, yd_bf, dL1[l]);
    attn_block(yd_bf, xe_bf, decCA[l], src_ids, 0);
    addln(yd, yd_bf, dL2[l]);
    ffn_block(yd_bf, decF[l]);
    addln(yd, yd_bf, dL3[l]);
  }
  // ---------------- tied logits: out = y @ emb^T (emb already [N,K]) ----------------
  cvt(emb, wbuf, (long long)cV * cD);
  gemm(yd_bf, cD, 0, wbuf, cD, 0,
       (float*)d_out, cV, 0, nullptr, 0, nullptr,
       cNT, cV, cD, 1, 1.0f, 0, 1, 0);
}
